// SinkhornCausalAttention_87033217286670
// MI455X (gfx1250) — compile-verified, matrix operand order checked by reading, not run on hardware
//
#include <hip/hip_runtime.h>
#include <math.h>

// Sinkhorn causal attention, CDNA5 (gfx1250) WMMA implementation.
// b=8, h=8, t=8192, dh=64, buckets=128, bsz=64, bh=64.

#define B_    8
#define H_    8
#define T_    8192
#define DH_   64
#define BKT_  128
#define BSZ_  64
#define BH_   64
#define HH_   4      // h/2: heads >= HH_ are the rotated half
#define SHIFT_ 63    // bsz - 1

typedef __attribute__((ext_vector_type(16))) _Float16 v16h;
typedef __attribute__((ext_vector_type(8)))  float    v8f;

struct f4 { float x, y, z, w; };

// K-index map for 16-bit WMMA A/B operands (wave32):
// lanes 0-15: halves 0..7 -> K 0..7,  halves 8..15 -> K 16..23
// lanes 16-31: halves 0..7 -> K 8..15, halves 8..15 -> K 24..31
__device__ __forceinline__ int kmap16(int lane, int e) {
  return e + ((e >> 3) << 3) + ((lane >> 4) << 3);
}

// ---------------------------------------------------------------------------
// Kernel A: cumulative-average routing features.
// x[bh, u, 0:64]  = cumavg of (rotated) k over [0, u*64] inclusive
// x[bh, u, 64:128]= (rotated) k at t = u*64
// ---------------------------------------------------------------------------
__global__ void sinkhorn_cumavg_kernel(const float* __restrict__ k,
                                       float* __restrict__ x) {
  int bh = blockIdx.x;          // 0..63
  int c  = threadIdx.x;         // 0..63 channel
  int hIdx = bh & (H_ - 1);
  bool rot = hIdx >= HH_;
  const float* kp = k + (size_t)bh * T_ * DH_;
  float sum = 0.f;
  for (int u = 0; u < BKT_; ++u) {
    int tstart = u * BSZ_;
    int s0 = rot ? ((tstart + SHIFT_) & (T_ - 1)) : tstart;
    float kv = kp[(size_t)s0 * DH_ + c];
    sum += kv;
    float* xr = x + ((size_t)bh * BKT_ + u) * (2 * DH_);
    xr[c]        = sum / (float)(tstart + 1);
    xr[DH_ + c]  = kv;
    for (int s = tstart + 1; s < tstart + BSZ_; ++s) {
      int ss = rot ? ((s + SHIFT_) & (T_ - 1)) : s;
      sum += kp[(size_t)ss * DH_ + c];
    }
  }
}

// ---------------------------------------------------------------------------
// Kernel B: routing logits -> masked softmax -> top-1 (index + weight).
// ---------------------------------------------------------------------------
__global__ void sinkhorn_route_kernel(const float* __restrict__ x,
                                      const float* __restrict__ sortW,
                                      int* __restrict__ idxArr,
                                      float* __restrict__ wArr) {
  __shared__ float sx[2 * DH_];
  __shared__ float sl[BKT_ + 1];
  int bid  = blockIdx.x;          // bh*128 + u
  int u    = bid & (BKT_ - 1);
  int bh   = bid >> 7;
  int hIdx = bh & (H_ - 1);
  int tid  = threadIdx.x;

  if (tid < 2 * DH_) sx[tid] = x[(size_t)bid * (2 * DH_) + tid];
  __syncthreads();

  if (tid <= BKT_) {
    const float* Wp = sortW + (size_t)hIdx * (2 * DH_) * (BKT_ + 1) + tid;
    float acc = 0.f;
    for (int dd = 0; dd < 2 * DH_; ++dd)
      acc = fmaf(sx[dd], Wp[(size_t)dd * (BKT_ + 1)], acc);
    sl[tid] = acc > 0.f ? acc : 0.01f * acc;   // leaky_relu
  }
  __syncthreads();

  if (tid == 0) {
    int best = 0; float wv = 0.f;
    if (u >= 1) {
      float m = -3.4e38f;
      for (int c2 = 0; c2 <= u; ++c2) m = fmaxf(m, sl[c2]);
      float s = 0.f;
      for (int c2 = 0; c2 <= u; ++c2) s += __expf(sl[c2] - m);
      float inv = 1.f / s;
      float bp = -1.f;
      for (int c2 = 0; c2 < u; ++c2) {        // strict lower triangle
        float p = __expf(sl[c2] - m) * inv;
        if (p > bp) { bp = p; best = c2; }
      }
      wv = bp;
    }
    idxArr[bid] = best;
    wArr[bid]   = wv;
  }
}

// ---------------------------------------------------------------------------
// Kernel C: bucketed attention. One block per (bh, bucket), 256 threads.
//   dots = Q(64x64) @ K2^T  -> 64x128  (64 WMMAs)
//   masked softmax per row
//   out  = P(64x128) @ V2   -> 64x64   (64 WMMAs)
// LDS (40KB): stage1 sQ[0..8K) sK[8K..24K) sS[24K..40K);
//             stage2 sVT overlays [0..16K).
// ---------------------------------------------------------------------------
__global__ __launch_bounds__(256) void sinkhorn_attn_kernel(
    const float* __restrict__ q, const float* __restrict__ k,
    const float* __restrict__ v, const float* __restrict__ nullK,
    const float* __restrict__ nullV, const int* __restrict__ idxArr,
    const float* __restrict__ wArr, float* __restrict__ out) {
  __shared__ __align__(16) char smem[40960];
  _Float16* sQ  = (_Float16*)(smem);          // 64x64   (stage 1)
  _Float16* sK  = (_Float16*)(smem + 8192);   // 128x64  (stage 1)
  _Float16* sS  = (_Float16*)(smem + 24576);  // 64x128 scores -> probs
  _Float16* sVT = (_Float16*)(smem);          // 64x128  (stage 2, V2^T)

  int bid  = blockIdx.x;
  int u    = bid & (BKT_ - 1);
  int bh   = bid >> 7;
  int hIdx = bh & (H_ - 1);
  bool rot = hIdx >= HH_;
  int tid  = threadIdx.x;
  int lane = tid & 31;
  int wave = tid >> 5;

  size_t base = (size_t)bh * T_ * DH_;
  int   ridx = idxArr[bid];
  float rw   = wArr[bid];

  // ---- stage Q and K2 into LDS as f16 (float4 global loads) ----
  for (int m = tid; m < (BSZ_ * DH_) / 4; m += 256) {
    int idx = m * 4;
    int i = idx >> 6, c = idx & 63;
    int tq = u * BSZ_ + i;
    int ts = rot ? ((tq + SHIFT_) & (T_ - 1)) : tq;

    f4 qa = *(const f4*)(q + base + (size_t)ts * DH_ + c);
    sQ[idx + 0] = (_Float16)qa.x; sQ[idx + 1] = (_Float16)qa.y;
    sQ[idx + 2] = (_Float16)qa.z; sQ[idx + 3] = (_Float16)qa.w;

    f4 ka = *(const f4*)(k + base + (size_t)ts * DH_ + c);
    _Float16* kd = sK + (BSZ_ + i) * DH_ + c;
    kd[0] = (_Float16)ka.x; kd[1] = (_Float16)ka.y;
    kd[2] = (_Float16)ka.z; kd[3] = (_Float16)ka.w;

    f4 kr;
    if (ridx == 0) {
      kr = *(const f4*)(nullK + hIdx * DH_ + c);   // null bucket row
    } else {
      int tr  = (ridx - 1) * BSZ_ + i;
      int trs = rot ? ((tr + SHIFT_) & (T_ - 1)) : tr;
      kr = *(const f4*)(k + base + (size_t)trs * DH_ + c);
    }
    _Float16* rd = sK + i * DH_ + c;
    rd[0] = (_Float16)(rw * kr.x); rd[1] = (_Float16)(rw * kr.y);
    rd[2] = (_Float16)(rw * kr.z); rd[3] = (_Float16)(rw * kr.w);
  }
  __syncthreads();

  // ---- dots = Q @ K2^T : 4x8 tiles of 16x16, K=64 in 2 steps ----
#pragma unroll
  for (int it = 0; it < 4; ++it) {
    int tt = wave + it * 8;               // uniform per wave, known trip count
    int mi = tt & 3, nj = tt >> 2;
    int rowA = mi * 16 + (lane & 15);
    int rowB = nj * 16 + (lane & 15);
    v8f acc = {0.f, 0.f, 0.f, 0.f, 0.f, 0.f, 0.f, 0.f};
#pragma unroll
    for (int k0 = 0; k0 < DH_; k0 += 32) {
      v16h a, bm;
#pragma unroll
      for (int e = 0; e < 16; ++e) {
        int kk = k0 + kmap16(lane, e);
        a[e]  = sQ[rowA * DH_ + kk];
        bm[e] = sK[rowB * DH_ + kk];
      }
      acc = __builtin_amdgcn_wmma_f32_16x16x32_f16(
          false, a, false, bm, (short)0, acc, false, false);
    }
#pragma unroll
    for (int r = 0; r < 8; ++r) {
      int row = mi * 16 + r + ((lane >> 4) << 3);
      int col = nj * 16 + (lane & 15);
      sS[row * (2 * BSZ_) + col] = (_Float16)acc[r];
    }
  }
  __syncthreads();

  // ---- masked softmax, one thread per row ----
  if (tid < BSZ_) {
    const float scale = 0.044194173824159216f;   // (h*dh)^-0.5 = 512^-0.5
    int i = tid;
    bool sp = rot && (u == BKT_ - 1) && (i >= 1);  // special last-bucket mask
    float m = -3.4e38f;
    for (int j = 0; j < 2 * BSZ_; ++j) {
      bool ok = (j < BSZ_ || (j - BSZ_) <= i) && !(sp && j <= BSZ_);
      if (ok) m = fmaxf(m, scale * (float)sS[i * (2 * BSZ_) + j]);
    }
    float sum = 0.f;
    for (int j = 0; j < 2 * BSZ_; ++j) {
      bool ok = (j < BSZ_ || (j - BSZ_) <= i) && !(sp && j <= BSZ_);
      float e = ok ? __expf(scale * (float)sS[i * (2 * BSZ_) + j] - m) : 0.f;
      sum += e;
      sS[i * (2 * BSZ_) + j] = (_Float16)e;
    }
    float inv = 1.f / sum;
    for (int j = 0; j < 2 * BSZ_; ++j)
      sS[i * (2 * BSZ_) + j] =
          (_Float16)((float)sS[i * (2 * BSZ_) + j] * inv);
  }
  __syncthreads();

  // ---- stage V2^T (channel-major, 64x128) overlaying sQ/sK ----
  for (int m = tid; m < (2 * BSZ_ * DH_) / 4; m += 256) {
    int idx = m * 4;
    int j = idx >> 6, c = idx & 63;     // c multiple of 4
    f4 va;
    float sc = 1.f;
    if (j < BSZ_) {                               // routed half
      sc = rw;
      if (ridx == 0) {
        va = *(const f4*)(nullV + hIdx * DH_ + c);
      } else {
        int tr  = (ridx - 1) * BSZ_ + j;
        int trs = rot ? ((tr + SHIFT_) & (T_ - 1)) : tr;
        va = *(const f4*)(v + base + (size_t)trs * DH_ + c);
      }
    } else {                                      // current bucket
      int tq = u * BSZ_ + (j - BSZ_);
      int ts = rot ? ((tq + SHIFT_) & (T_ - 1)) : tq;
      va = *(const f4*)(v + base + (size_t)ts * DH_ + c);
    }
    sVT[(c + 0) * (2 * BSZ_) + j] = (_Float16)(sc * va.x);
    sVT[(c + 1) * (2 * BSZ_) + j] = (_Float16)(sc * va.y);
    sVT[(c + 2) * (2 * BSZ_) + j] = (_Float16)(sc * va.z);
    sVT[(c + 3) * (2 * BSZ_) + j] = (_Float16)(sc * va.w);
  }
  __syncthreads();

  // ---- out = P @ V2 : 4x4 tiles of 16x16, K=128 in 4 steps ----
#pragma unroll
  for (int it = 0; it < 2; ++it) {
    int tt = wave + it * 8;
    int mi = tt & 3, nj = tt >> 2;
    int rowA = mi * 16 + (lane & 15);
    int rowB = nj * 16 + (lane & 15);   // channel index
    v8f acc = {0.f, 0.f, 0.f, 0.f, 0.f, 0.f, 0.f, 0.f};
#pragma unroll
    for (int k0 = 0; k0 < 2 * BSZ_; k0 += 32) {
      v16h a, bm;
#pragma unroll
      for (int e = 0; e < 16; ++e) {
        int kk = k0 + kmap16(lane, e);
        a[e]  = sS[rowA * (2 * BSZ_) + kk];
        bm[e] = sVT[rowB * (2 * BSZ_) + kk];
      }
      acc = __builtin_amdgcn_wmma_f32_16x16x32_f16(
          false, a, false, bm, (short)0, acc, false, false);
    }
#pragma unroll
    for (int r = 0; r < 8; ++r) {
      int i = mi * 16 + r + ((lane >> 4) << 3);
      int e = nj * 16 + (lane & 15);
      int tq = u * BSZ_ + i;
      int to = rot ? ((tq + SHIFT_) & (T_ - 1)) : tq;  // un-rotate output
      out[base + (size_t)to * DH_ + e] = acc[r];
    }
  }
}

// ---------------------------------------------------------------------------
extern "C" void kernel_launch(void* const* d_in, const int* in_sizes, int n_in,
                              void* d_out, int out_size, void* d_ws,
                              size_t ws_size, hipStream_t stream) {
  const float* q     = (const float*)d_in[0];
  const float* k     = (const float*)d_in[1];
  const float* v     = (const float*)d_in[2];
  const float* sortW = (const float*)d_in[3];
  const float* nullK = (const float*)d_in[4];
  const float* nullV = (const float*)d_in[5];
  float* out = (float*)d_out;

  char* ws = (char*)d_ws;
  float* x      = (float*)ws;                       // 64*128*128 f32 = 4 MiB
  int*   idxArr = (int*)(ws + 4194304);             // 8192 ints
  float* wArr   = (float*)(ws + 4194304 + 32768);   // 8192 floats

  sinkhorn_cumavg_kernel<<<BH_, DH_, 0, stream>>>(k, x);
  sinkhorn_route_kernel<<<BH_ * BKT_, 256, 0, stream>>>(x, sortW, idxArr, wArr);
  sinkhorn_attn_kernel<<<BH_ * BKT_, 256, 0, stream>>>(q, k, v, nullK, nullV,
                                                       idxArr, wArr, out);
}
